// RNN_22282290332161
// MI455X (gfx1250) — matrix operand order repaired and checked by reading, not compiled, not run on hardware
//
#include <hip/hip_runtime.h>

#define INPUT 300
#define HDIM  128
#define BATCH 256
#define TLEN  512
#define MTOT  (BATCH * TLEN)

typedef __bf16 bf16_t;
typedef bf16_t v16bf __attribute__((ext_vector_type(16)));
typedef float  v8f   __attribute__((ext_vector_type(8)));

union Frag16 { v16bf v; unsigned int u[8]; };
union AccF   { v8f   v; float        f[8]; };

__device__ __forceinline__ unsigned int pack_bf16x2(float lo, float hi) {
  union { float f; unsigned int u; } a, b;
  a.f = lo; b.f = hi;
  unsigned int ul = (a.u + 0x7FFFu + ((a.u >> 16) & 1u)) >> 16;
  unsigned int uh = (b.u + 0x7FFFu + ((b.u >> 16) & 1u)) >> 16;
  return (ul & 0xFFFFu) | (uh << 16);
}
__device__ __forceinline__ unsigned short f2bf(float f) {
  union { float f; unsigned int u; } a; a.f = f;
  return (unsigned short)((a.u + 0x7FFFu + ((a.u >> 16) & 1u)) >> 16);
}
__device__ __forceinline__ float bf2f(unsigned short h) {
  union { unsigned int u; float f; } a; a.u = ((unsigned int)h) << 16;
  return a.f;
}
__device__ __forceinline__ float fast_tanh(float x) {
#if __has_builtin(__builtin_amdgcn_tanhf)
  return __builtin_amdgcn_tanhf(x);     // CDNA5 V_TANH_F32 trans op
#else
  return tanhf(x);
#endif
}

// ---------------------------------------------------------------------------
// Phase 1: xp[m, n] = sum_k x[m, k] * W_ih[n, k] + b_ih[n] + b_hh[n]
// M = B*T = 131072; one workgroup = 128x128 tile, 8 waves. Double-buffered
// LDS: global loads for chunk k+1 overlap the WMMAs of chunk k; one barrier
// per chunk. HBM-bound: 157 MB of x -> ~7us floor at 23.3 TB/s.
// ---------------------------------------------------------------------------
__global__ __launch_bounds__(256) void rnn_xproj_kernel(
    const float* __restrict__ x, const float* __restrict__ W_ih,
    const float* __restrict__ b_ih, const float* __restrict__ b_hh,
    float* __restrict__ xp)
{
  __shared__ unsigned int a_lds[2][128 * 17];  // 128 rows x 32 bf16 (+pad)
  __shared__ unsigned int b_lds[2][128 * 17];
  __shared__ float bias[HDIM];

  const int tid  = threadIdx.x;
  const int lane = tid & 31;
  const int wv   = tid >> 5;
  const int half = lane >> 4;
  const int l16  = lane & 15;
  const long m0  = (long)blockIdx.x * 128;

  if (tid < HDIM) bias[tid] = b_ih[tid] + b_hh[tid];

  AccF acc[8];
#pragma unroll
  for (int nt = 0; nt < 8; ++nt)
#pragma unroll
    for (int r = 0; r < 8; ++r) acc[nt].f[r] = 0.0f;

  float4 rA[4], rB[4];
  auto load_chunk = [&](int kc) {
    const int k0 = kc * 32;
#pragma unroll
    for (int it = 0; it < 4; ++it) {
      int idx = tid + it * 256;          // 0..1023 float4 slots
      int row = idx >> 3;
      int cv  = idx & 7;
      float4 zA = float4{0.f, 0.f, 0.f, 0.f};
      float4 zB = float4{0.f, 0.f, 0.f, 0.f};
      if (k0 + cv * 4 + 4 <= INPUT) {    // rows are 1200B -> float4 aligned
        zA = *(const float4*)(x    + (m0 + row) * INPUT + k0 + cv * 4);
        zB = *(const float4*)(W_ih + (long)row  * INPUT + k0 + cv * 4);
      }
      rA[it] = zA; rB[it] = zB;
    }
  };
  load_chunk(0);

  for (int kc = 0; kc < 10; ++kc) {      // K = 300 in 10 chunks of 32 (padded)
    unsigned int (&A)[128 * 17]  = a_lds[kc & 1];
    unsigned int (&Bm)[128 * 17] = b_lds[kc & 1];
#pragma unroll
    for (int it = 0; it < 4; ++it) {
      int idx = tid + it * 256;
      int row = idx >> 3, cv = idx & 7;
      A[row * 17 + cv * 2    ]  = pack_bf16x2(rA[it].x, rA[it].y);
      A[row * 17 + cv * 2 + 1]  = pack_bf16x2(rA[it].z, rA[it].w);
      Bm[row * 17 + cv * 2    ] = pack_bf16x2(rB[it].x, rB[it].y);
      Bm[row * 17 + cv * 2 + 1] = pack_bf16x2(rB[it].z, rB[it].w);
    }
    __syncthreads();
    if (kc + 1 < 10) load_chunk(kc + 1); // issue next chunk's HBM loads now

    Frag16 af;                           // A: this wave's 16 M-rows x 32 K
#pragma unroll
    for (int j = 0; j < 8; ++j)
      af.u[j] = A[(16 * wv + l16) * 17 + (j >> 2) * 8 + half * 4 + (j & 3)];
#pragma unroll
    for (int nt = 0; nt < 8; ++nt) {
      Frag16 bfr;
#pragma unroll
      for (int j = 0; j < 8; ++j)
        bfr.u[j] = Bm[(nt * 16 + l16) * 17 + (j >> 2) * 8 + half * 4 + (j & 3)];
      acc[nt].v = __builtin_amdgcn_wmma_f32_16x16x32_bf16(
          false, af.v, false, bfr.v, (short)0, acc[nt].v, false, false);
    }
    // no second barrier: double buffering orders reuse via next iteration's barrier
  }

#pragma unroll
  for (int nt = 0; nt < 8; ++nt) {
    int n = nt * 16 + l16;
    float bb = bias[n];
#pragma unroll
    for (int r = 0; r < 8; ++r) {
      long row = m0 + 16 * wv + r + 8 * half;
      xp[row * HDIM + n] = acc[nt].f[r] + bb;
    }
  }
}

// ---------------------------------------------------------------------------
// Phase 2: h_t = tanh(xp[:,t,:] + h_{t-1} @ W_hh^T); out = h_T @ fc_w^T + fc_b
// 16 independent workgroups x 16 batch rows. W_hh B-fragments pinned in VGPRs
// for all 512 steps; h in LDS (bf16). xp loads software-pipelined distance 2
// (xp fits in 192MB L2, just written by phase 1). tanh via V_TANH_F32.
// ---------------------------------------------------------------------------
__global__ __launch_bounds__(256) void rnn_recur_kernel(
    const float* __restrict__ xp, const float* __restrict__ W_hh,
    const float* __restrict__ fc_w, const float* __restrict__ fc_b,
    float* __restrict__ out)
{
  __shared__ unsigned int whh[128 * 65];   // [n][k] bf16 (+pad)
  __shared__ unsigned int hbuf[16 * 65];   // [m][n] bf16 (+pad)

  const int tid  = threadIdx.x;
  const int lane = tid & 31;
  const int wv   = tid >> 5;               // n-tile owned by this wave
  const int half = lane >> 4;
  const int l16  = lane & 15;
  const int b0   = blockIdx.x * 16;

#pragma unroll
  for (int it = 0; it < 16; ++it) {        // W_hh -> bf16 LDS, [n][k]
    int idx = tid + it * 256;
    int n = idx >> 5, cv = idx & 31;
    float4 v = *(const float4*)(W_hh + n * HDIM + cv * 4);
    whh[n * 65 + cv * 2    ] = pack_bf16x2(v.x, v.y);
    whh[n * 65 + cv * 2 + 1] = pack_bf16x2(v.z, v.w);
  }
  for (int i = tid; i < 16 * 65; i += 256) hbuf[i] = 0u;   // h0 = 0
  __syncthreads();

  Frag16 bfr[4];                           // W_hh fragments: resident all loop
#pragma unroll
  for (int kc = 0; kc < 4; ++kc)
#pragma unroll
    for (int j = 0; j < 8; ++j)
      bfr[kc].u[j] = whh[(wv * 16 + l16) * 65 + kc * 16 + (j >> 2) * 8 + half * 4 + (j & 3)];

  const int n_col = wv * 16 + l16;
  const float* xrow[8];
#pragma unroll
  for (int r = 0; r < 8; ++r)
    xrow[r] = xp + (long)(b0 + r + 8 * half) * TLEN * HDIM + n_col;

  float xv0[8], xv1[8];
#pragma unroll
  for (int r = 0; r < 8; ++r) {            // prime pipeline: t=0, t=1
    xv0[r] = xrow[r][0 * HDIM];
    xv1[r] = xrow[r][1 * HDIM];
  }

  auto step = [&](int t, float (&buf)[8]) {
    __syncthreads();                       // h_{t-1} visible
    Frag16 af[4];
#pragma unroll
    for (int kc = 0; kc < 4; ++kc)
#pragma unroll
      for (int j = 0; j < 8; ++j)
        af[kc].u[j] = hbuf[l16 * 65 + kc * 16 + (j >> 2) * 8 + half * 4 + (j & 3)];

    AccF acc0, acc1;                       // two chains -> halve WMMA dep chain
#pragma unroll
    for (int r = 0; r < 8; ++r) { acc0.f[r] = 0.0f; acc1.f[r] = 0.0f; }
    acc0.v = __builtin_amdgcn_wmma_f32_16x16x32_bf16(false, af[0].v, false, bfr[0].v, (short)0, acc0.v, false, false);
    acc1.v = __builtin_amdgcn_wmma_f32_16x16x32_bf16(false, af[1].v, false, bfr[1].v, (short)0, acc1.v, false, false);
    acc0.v = __builtin_amdgcn_wmma_f32_16x16x32_bf16(false, af[2].v, false, bfr[2].v, (short)0, acc0.v, false, false);
    acc1.v = __builtin_amdgcn_wmma_f32_16x16x32_bf16(false, af[3].v, false, bfr[3].v, (short)0, acc1.v, false, false);

    __syncthreads();                       // all reads of h done before overwrite
    float cur[8];
#pragma unroll
    for (int r = 0; r < 8; ++r) cur[r] = buf[r];
    if (t + 2 < TLEN) {                    // prefetch xp for t+2 (distance 2)
#pragma unroll
      for (int r = 0; r < 8; ++r) buf[r] = xrow[r][(long)(t + 2) * HDIM];
    }
#pragma unroll
    for (int r = 0; r < 8; ++r) {
      float hv = fast_tanh(acc0.f[r] + acc1.f[r] + cur[r]);
      ((unsigned short*)hbuf)[(r + 8 * half) * 130 + n_col] = f2bf(hv);
    }
  };

  for (int tt = 0; tt < TLEN; tt += 2) {   // two steps/iter keeps ping-pong
    step(tt,     xv0);                     // buffers constant-indexed (no spill)
    step(tt + 1, xv1);
  }
  __syncthreads();

  if (tid < 32) {                          // out[b0+m, c] = h_T . fc_w[c] + fc_b[c]
    int m = tid >> 1, c = tid & 1;
    float s = fc_b[c];
    for (int n = 0; n < HDIM; ++n) {
      unsigned int u = hbuf[m * 65 + (n >> 1)];
      unsigned short hb = (n & 1) ? (unsigned short)(u >> 16)
                                  : (unsigned short)(u & 0xFFFFu);
      s += bf2f(hb) * fc_w[c * HDIM + n];
    }
    out[(b0 + m) * 2 + c] = s;
  }
}

extern "C" void kernel_launch(void* const* d_in, const int* in_sizes, int n_in,
                              void* d_out, int out_size, void* d_ws, size_t ws_size,
                              hipStream_t stream) {
  const float* x    = (const float*)d_in[0];
  const float* W_ih = (const float*)d_in[1];
  const float* W_hh = (const float*)d_in[2];
  const float* b_ih = (const float*)d_in[3];
  const float* b_hh = (const float*)d_in[4];
  const float* fc_w = (const float*)d_in[5];
  const float* fc_b = (const float*)d_in[6];
  float* out = (float*)d_out;
  float* xp  = (float*)d_ws;   // [B*T, 128] f32 = 64 MB scratch

  rnn_xproj_kernel<<<MTOT / 128, 256, 0, stream>>>(x, W_ih, b_ih, b_hh, xp);
  rnn_recur_kernel<<<BATCH / 16, 256, 0, stream>>>(xp, W_hh, fc_w, fc_b, out);
}
